// Sae_20658792693955
// MI455X (gfx1250) — matrix operand order repaired and checked by reading, not compiled, not run on hardware
//
#include <hip/hip_runtime.h>
#include <cstdint>
#include <cstddef>

// ---------------- problem constants ----------------
#define NROW 4096      // batch N
#define DIN  1024      // d_in
#define NLAT 32768     // n_latents
#define K4   128       // 4*k (top-128; top-32 is its sorted prefix)
#define CAP  1024      // candidate capacity for threshold collection

// GEMM tiling
#define ROWS_PER_WG 64
#define COLS_PER_WG 128
#define KSLICE      128
#define NSLICE      (DIN / KSLICE)
#define LSTRIDE     136   // 128 + 8 pad halves -> 272B row stride (17 dwords x 16) = conflict-free

// CDNA5 async global->LDS copy (ASYNCcnt-tracked, no VGPR data path)
#if defined(__has_builtin)
#  if __has_builtin(__builtin_amdgcn_global_load_async_to_lds_b128)
#    define USE_ASYNC_LDS 1
#  endif
#endif
#ifndef USE_ASYNC_LDS
#  define USE_ASYNC_LDS 0
#endif

typedef __attribute__((ext_vector_type(16))) _Float16 v16h;
typedef __attribute__((ext_vector_type(8)))  _Float16 v8h;
typedef __attribute__((ext_vector_type(8)))  float    v8f;
typedef __attribute__((ext_vector_type(4)))  int      v4i;

// ---------------- f32 -> f16 conversion ----------------
__global__ __launch_bounds__(256) void sae_cvt_x(const float* __restrict__ x,
                                                 const float* __restrict__ b_dec,
                                                 _Float16* __restrict__ xh) {
  for (size_t i = (size_t)blockIdx.x * 256 + threadIdx.x; i < (size_t)NROW * DIN;
       i += (size_t)gridDim.x * 256) {
    int d = (int)(i & (DIN - 1));
    xh[i] = (_Float16)(x[i] - b_dec[d]);   // fold encoder's (x - b_dec)
  }
}

__global__ __launch_bounds__(256) void sae_cvt_w(const float* __restrict__ w,
                                                 _Float16* __restrict__ wh) {
  for (size_t i = (size_t)blockIdx.x * 256 + threadIdx.x; i < (size_t)NLAT * DIN;
       i += (size_t)gridDim.x * 256) {
    wh[i] = (_Float16)w[i];
  }
}

// ---------------- encoder GEMM: pre = relu(xh @ whT + b_enc) ----------------
// Workgroup: 256 threads = 8 wave32s. Tile: 64 rows x 128 latent cols.
// Each wave owns one 16-col stripe and 4 row tiles, so every B fragment
// (W_enc, streamed from L2; f16 W = 64 MB resident in 192 MB L2) feeds 4 wmma.
// x K-slices are staged into double-buffered LDS via the CDNA5 async
// global->LDS path: the DMA copy of slice s+1 overlaps the 16 wmma of slice
// s with zero VGPR cost (the register-prefetch variant spilled to scratch).
__global__ __launch_bounds__(256) void sae_enc_gemm(const _Float16* __restrict__ xh,
                                                    const _Float16* __restrict__ wh,
                                                    const float* __restrict__ b_enc,
                                                    float* __restrict__ pre) {
  __shared__ _Float16 sx[2][ROWS_PER_WG * LSTRIDE];   // 2 x 17 KB
  const int tid = threadIdx.x;
  const int colBlock = blockIdx.x * COLS_PER_WG;
  const int rowBlock = blockIdx.y * ROWS_PER_WG;
  const int wave = tid >> 5;
  const int lane = tid & 31;
  const int r    = lane & 15;   // A row within tile / B column within stripe
  const int half = lane >> 4;

  // staging assignment: 64 rows x 128 halves = 1024 16B chunks, 4 per thread
  const int strow = tid >> 4;
  const int stcs  = tid & 15;
  const _Float16* xsrc = xh + (size_t)rowBlock * DIN + (size_t)strow * DIN + stcs * 8;
  _Float16* sdst0 = &sx[0][strow * LSTRIDE + stcs * 8];
  _Float16* sdst1 = &sx[1][strow * LSTRIDE + stcs * 8];
  const size_t xrowstep = (size_t)16 * DIN;     // +256 threads -> +16 rows
  const int    srowstep = 16 * LSTRIDE;

  // copy one K-slice chunk set: 4 x 16B per thread, global -> LDS
  auto stage_slice = [&](const _Float16* gsrc, _Float16* ldst) {
#if USE_ASYNC_LDS
#pragma unroll
    for (int j = 0; j < 4; ++j)
      __builtin_amdgcn_global_load_async_to_lds_b128(
          (__attribute__((address_space(1))) v4i*)(gsrc + xrowstep * j),
          (__attribute__((address_space(3))) v4i*)(ldst + srowstep * j),
          /*offset=*/0, /*cpol=*/0);
#else
    uint4 tmp[4];
#pragma unroll
    for (int j = 0; j < 4; ++j) tmp[j] = *(const uint4*)(gsrc + xrowstep * j);
#pragma unroll
    for (int j = 0; j < 4; ++j) *(uint4*)(ldst + srowstep * j) = tmp[j];
#endif
  };
  auto stage_wait = []() {
#if USE_ASYNC_LDS
    asm volatile("s_wait_asynccnt 0x0" ::: "memory");   // ISA 15.5, SOPP op 74
#endif
  };

  // stage slice 0
  stage_slice(xsrc, sdst0);
  stage_wait();
  __syncthreads();

  // B fragment source: lane holds latent column (colBlock + wave*16 + r),
  // 16 contiguous K values starting at half*16 (contiguous K order, same as A)
  const _Float16* bp = wh + (size_t)(colBlock + wave * 16 + r) * DIN + half * 16;

  // A fragment load (contiguous K: lane half selects K 16*half .. 16*half+15, matching B)
  auto loadA = [&](const _Float16* base, int rt, int j) -> v16h {
    const _Float16* ap = base + (rt * 16 + r) * LSTRIDE + j * 32 + half * 16;
    v8h lo = *(const v8h*)(ap);
    v8h hi = *(const v8h*)(ap + 8);
    return __builtin_shufflevector(lo, hi, 0, 1, 2, 3, 4, 5, 6, 7,
                                   8, 9, 10, 11, 12, 13, 14, 15);
  };

  v8f acc[4] = {};

#pragma unroll 1
  for (int s = 0; s < NSLICE; ++s) {
    const _Float16* cur = sx[s & 1];

    // kick off the async copy of slice s+1 into the other buffer. Safe: the
    // barrier ending iteration s-1 proves every wave finished reading that
    // buffer (it held slice s-1).
    if (s < NSLICE - 1)
      stage_slice(xsrc + (size_t)(s + 1) * KSLICE, ((s + 1) & 1) ? sdst1 : sdst0);

    // preload this slice's 4 B fragments into distinct registers so the
    // global loads pipeline instead of each wmma draining loadcnt to 0
    const _Float16* bs = bp + s * KSLICE;
    v16h bfrag[4];
#pragma unroll
    for (int j = 0; j < 4; ++j) bfrag[j] = *(const v16h*)(bs + j * 32);

    // 4 k-steps x 4 row tiles = 16 wmma per slice; acc chains are per row
    // tile (4 independent chains -> good XDL pipelining)
#pragma unroll
    for (int j = 0; j < 4; ++j) {
#pragma unroll
      for (int rt = 0; rt < 4; ++rt) {
        v16h a = loadA(cur, rt, j);
        acc[rt] = __builtin_amdgcn_wmma_f32_16x16x32_f16(
            /*neg_a=*/false, a, /*neg_b=*/false, bfrag[j],
            /*c_mod=*/(short)0, acc[rt], /*reuse_a=*/false, /*reuse_b=*/false);
      }
    }

    if (s < NSLICE - 1) {
      stage_wait();      // this wave's async transfers into LDS complete
      __syncthreads();   // all waves' transfers visible -> slice s+1 ready
    }
  }

  // epilogue: + b_enc, relu, store. C layout: VGPR i -> row (half*8 + i), col = lane&15
  const int col = colBlock + wave * 16 + r;
  const float be = b_enc[col];
#pragma unroll
  for (int rt = 0; rt < 4; ++rt) {
    int mbase = rowBlock + rt * 16 + (half << 3);
#pragma unroll
    for (int i = 0; i < 8; ++i) {
      float v = acc[rt][i] + be;
      v = v > 0.f ? v : 0.f;
      pre[(size_t)(mbase + i) * NLAT + col] = v;
    }
  }
}

// ---------------- per-row top-128 via 4-pass radix select ----------------
// One 256-thread block per row. Values are relu'd (>=0) so the f32 bit pattern
// is monotone. Also produces the row L1 sum on pass 0.
__global__ __launch_bounds__(256) void sae_topk(const float* __restrict__ pre,
                                                float* __restrict__ ta4,
                                                int* __restrict__ ti4,
                                                float* __restrict__ row_l1) {
  const int row = blockIdx.x;
  const int tid = threadIdx.x;
  const float* p = pre + (size_t)row * NLAT;

  __shared__ unsigned hist[256];
  __shared__ unsigned long long comp[CAP];
  __shared__ float red[256];
  __shared__ unsigned sh_prefix;
  __shared__ int sh_need;
  __shared__ int cnt;

  unsigned prefix = 0;
  int need = K4;

  for (int pass = 0; pass < 4; ++pass) {
    hist[tid] = 0;
    __syncthreads();
    float lsum = 0.f;
    for (int i = tid; i < NLAT; i += 256) {
      float v = p[i];
      unsigned key = __float_as_uint(v);
      if (pass == 0) {
        lsum += v;
        atomicAdd(&hist[key >> 24], 1u);
      } else if ((key >> (32 - 8 * pass)) == prefix) {
        atomicAdd(&hist[(key >> (24 - 8 * pass)) & 255u], 1u);
      }
    }
    if (pass == 0) {   // row L1 = sum of relu'd pre-acts
      red[tid] = lsum;
      __syncthreads();
      for (int s = 128; s > 0; s >>= 1) {
        if (tid < s) red[tid] += red[tid + s];
        __syncthreads();
      }
      if (tid == 0) row_l1[row] = red[0];
    }
    __syncthreads();
    if (tid == 0) {
      int c = 0, b = 255;
      for (; b >= 0; --b) {
        int nc = c + (int)hist[b];
        if (nc >= need) break;
        c = nc;
      }
      if (b < 0) b = 0;
      sh_need = need - c;
      sh_prefix = (prefix << 8) | (unsigned)b;
    }
    __syncthreads();
    prefix = sh_prefix;
    need = sh_need;
    __syncthreads();
  }

  // collect everything >= threshold; pack (key << 32) | (~idx) for desc-value/asc-idx sort
  if (tid == 0) cnt = 0;
  __syncthreads();
  const unsigned thr = prefix;
  for (int i = tid; i < NLAT; i += 256) {
    unsigned key = __float_as_uint(p[i]);
    if (key >= thr) {
      int pos = atomicAdd(&cnt, 1);
      if (pos < CAP)
        comp[pos] = ((unsigned long long)key << 32) |
                    (unsigned long long)(0xFFFFFFFFu - (unsigned)i);
    }
  }
  __syncthreads();
  int M = cnt < CAP ? cnt : CAP;
  for (int i = tid; i < CAP; i += 256)
    if (i >= M) comp[i] = 0ull;
  __syncthreads();

  // bitonic sort, descending
  for (int k = 2; k <= CAP; k <<= 1) {
    for (int j = k >> 1; j > 0; j >>= 1) {
      for (int t = tid; t < CAP / 2; t += 256) {
        int i = ((t / j) * (2 * j)) + (t % j);
        int ixj = i + j;
        unsigned long long a = comp[i], b = comp[ixj];
        bool desc = ((i & k) == 0);
        if (desc ? (a < b) : (a > b)) { comp[i] = b; comp[ixj] = a; }
      }
      __syncthreads();
    }
  }

  if (tid < K4) {
    unsigned long long c = comp[tid];
    float v = 0.f;
    int idx = 0;
    if (tid < M) {
      v = __uint_as_float((unsigned)(c >> 32));
      idx = (int)(0xFFFFFFFFu - (unsigned)(c & 0xFFFFFFFFull));
    }
    ta4[(size_t)row * K4 + tid] = v;
    ti4[(size_t)row * K4 + tid] = idx;
  }
}

// ---------------- sparse decode + per-row loss partials ----------------
__global__ __launch_bounds__(256) void sae_decode(const float* __restrict__ x,
                                                  const float* __restrict__ W_dec,
                                                  const float* __restrict__ b_dec,
                                                  const float* __restrict__ ta4,
                                                  const int* __restrict__ ti4,
                                                  float* __restrict__ sae,
                                                  float* __restrict__ row_se2,
                                                  float* __restrict__ row_se4,
                                                  float* __restrict__ row_sx2) {
  const int row = blockIdx.x;
  const int tid = threadIdx.x;
  __shared__ float sa[K4];
  __shared__ int si[K4];
  __shared__ float red[256];
  if (tid < K4) {
    sa[tid] = ta4[(size_t)row * K4 + tid];
    si[tid] = ti4[(size_t)row * K4 + tid];
  }
  __syncthreads();

  float a0 = 0, a1 = 0, a2 = 0, a3 = 0;      // top-32 accumulator
  for (int k = 0; k < 32; ++k) {
    float a = sa[k];
    const float* wr = W_dec + (size_t)si[k] * DIN;
    a0 += a * wr[tid]; a1 += a * wr[tid + 256];
    a2 += a * wr[tid + 512]; a3 += a * wr[tid + 768];
  }
  float b0 = a0, b1 = a1, b2 = a2, b3 = a3;  // continue to top-128
  for (int k = 32; k < K4; ++k) {
    float a = sa[k];
    const float* wr = W_dec + (size_t)si[k] * DIN;
    b0 += a * wr[tid]; b1 += a * wr[tid + 256];
    b2 += a * wr[tid + 512]; b3 += a * wr[tid + 768];
  }

  float se2 = 0, se4 = 0, sx2 = 0;
  float acc[4]  = {a0, a1, a2, a3};
  float acc4[4] = {b0, b1, b2, b3};
#pragma unroll
  for (int j = 0; j < 4; ++j) {
    int d = tid + 256 * j;
    float bd = b_dec[d];
    float xv = x[(size_t)row * DIN + d];
    float o = acc[j] + bd, o4 = acc4[j] + bd;
    sae[(size_t)row * DIN + d] = o;
    float e = o - xv, e4 = o4 - xv;
    se2 += e * e; se4 += e4 * e4; sx2 += xv * xv;
  }

  // three fixed-order block reductions (deterministic)
  float t_se2, t_se4, t_sx2;
  red[tid] = se2; __syncthreads();
  for (int s = 128; s > 0; s >>= 1) { if (tid < s) red[tid] += red[tid + s]; __syncthreads(); }
  t_se2 = red[0]; __syncthreads();
  red[tid] = se4; __syncthreads();
  for (int s = 128; s > 0; s >>= 1) { if (tid < s) red[tid] += red[tid + s]; __syncthreads(); }
  t_se4 = red[0]; __syncthreads();
  red[tid] = sx2; __syncthreads();
  for (int s = 128; s > 0; s >>= 1) { if (tid < s) red[tid] += red[tid + s]; __syncthreads(); }
  t_sx2 = red[0];
  if (tid == 0) { row_se2[row] = t_se2; row_se4[row] = t_se4; row_sx2[row] = t_sx2; }
}

// column sums of x -> per-dim (colsum)^2 for total-variance
__global__ __launch_bounds__(256) void sae_colsq(const float* __restrict__ x,
                                                 float* __restrict__ colsq) {
  int d = blockIdx.x * 256 + threadIdx.x;
  float s = 0.f;
  for (int n = 0; n < NROW; ++n) s += x[(size_t)n * DIN + d];
  colsq[d] = s * s;
}

// final scalar losses (single block, fixed-order => deterministic)
__global__ __launch_bounds__(256) void sae_finalize(const float* __restrict__ row_se2,
                                                    const float* __restrict__ row_se4,
                                                    const float* __restrict__ row_sx2,
                                                    const float* __restrict__ row_l1,
                                                    const float* __restrict__ colsq,
                                                    float* __restrict__ scal) {
  __shared__ float red[256];
  const int tid = threadIdx.x;
  float se2 = 0, se4 = 0, sx2 = 0, rec = 0, l1 = 0, csq = 0;
  for (int r = tid; r < NROW; r += 256) {
    float a = row_se2[r], b = row_se4[r], c = row_sx2[r];
    se2 += a; se4 += b; sx2 += c;
    rec += a / c;                    // mean_d(e^2)/mean_d(x^2) == se2/sx2
    l1 += row_l1[r] * rsqrtf(c);     // sum|pre| / ||x||
  }
  for (int d = tid; d < DIN; d += 256) csq += colsq[d];

  auto reduce = [&](float v) -> float {
    red[tid] = v; __syncthreads();
    for (int s = 128; s > 0; s >>= 1) { if (tid < s) red[tid] += red[tid + s]; __syncthreads(); }
    float r0 = red[0]; __syncthreads();
    return r0;
  };
  float t_se2 = reduce(se2), t_se4 = reduce(se4), t_sx2 = reduce(sx2);
  float t_rec = reduce(rec), t_l1 = reduce(l1), t_csq = reduce(csq);
  if (tid == 0) {
    float tv = t_sx2 - t_csq * (1.0f / NROW);   // sum((x-mean)^2)
    scal[0] = t_se2 / tv;                       // fvu
    scal[1] = 0.0f;                             // auxk_loss
    scal[2] = t_se4 / tv;                       // multi_topk_fvu
    scal[3] = t_l1 * (1.0f / NROW);             // l1_loss
    scal[4] = t_rec * (1.0f / NROW);            // reconstruction_loss
  }
}

// ---------------- host launcher ----------------
extern "C" void kernel_launch(void* const* d_in, const int* in_sizes, int n_in,
                              void* d_out, int out_size, void* d_ws, size_t ws_size,
                              hipStream_t stream) {
  (void)in_sizes; (void)n_in; (void)out_size; (void)ws_size;
  const float* x     = (const float*)d_in[0];
  const float* W_enc = (const float*)d_in[1];
  const float* b_enc = (const float*)d_in[2];
  const float* W_dec = (const float*)d_in[3];
  const float* b_dec = (const float*)d_in[4];

  // workspace layout (~584 MB)
  char* p = (char*)d_ws;
  _Float16* xh = (_Float16*)p;             p += (size_t)NROW * DIN * 2;   //   8 MB
  _Float16* wh = (_Float16*)p;             p += (size_t)NLAT * DIN * 2;   //  64 MB
  float* pre      = (float*)p;             p += (size_t)NROW * NLAT * 4;  // 512 MB
  float* row_l1   = (float*)p;             p += (size_t)NROW * 4;
  float* row_se2  = (float*)p;             p += (size_t)NROW * 4;
  float* row_se4  = (float*)p;             p += (size_t)NROW * 4;
  float* row_sx2  = (float*)p;             p += (size_t)NROW * 4;
  float* colsq    = (float*)p;             p += (size_t)DIN * 4;

  // output layout: sae_out | top_acts4 | top_idx4 | 5 scalars
  float* out  = (float*)d_out;
  float* sae  = out;
  float* ta4  = out + (size_t)NROW * DIN;
  int*   ti4  = (int*)(out + (size_t)NROW * DIN + (size_t)NROW * K4);
  float* scal = out + (size_t)NROW * DIN + (size_t)2 * NROW * K4;

  sae_cvt_x<<<2048, 256, 0, stream>>>(x, b_dec, xh);
  sae_cvt_w<<<8192, 256, 0, stream>>>(W_enc, wh);

  dim3 ggrid(NLAT / COLS_PER_WG, NROW / ROWS_PER_WG);   // (256, 64)
  sae_enc_gemm<<<ggrid, 256, 0, stream>>>(xh, wh, b_enc, pre);

  sae_topk<<<NROW, 256, 0, stream>>>(pre, ta4, ti4, row_l1);

  sae_decode<<<NROW, 256, 0, stream>>>(x, W_dec, b_dec, ta4, ti4, sae,
                                       row_se2, row_se4, row_sx2);

  sae_colsq<<<DIN / 256, 256, 0, stream>>>(x, colsq);

  sae_finalize<<<1, 256, 0, stream>>>(row_se2, row_se4, row_sx2, row_l1, colsq, scal);
}
